// Dual_Attn_75917841924508
// MI455X (gfx1250) — matrix operand order repaired and checked by reading, not compiled, not run on hardware
//
#include <hip/hip_runtime.h>

#define C_IN 512
#define DK   64
#define NTOK 4096
#define BATCH 4

typedef _Float16 half_t;
typedef __attribute__((ext_vector_type(8)))  _Float16 v8h;
typedef __attribute__((ext_vector_type(16))) _Float16 v16h;
typedef __attribute__((ext_vector_type(8)))  float    v8f;
typedef __attribute__((ext_vector_type(4)))  unsigned int u32x4;
typedef __attribute__((ext_vector_type(8)))  int      i32x8;
typedef __attribute__((ext_vector_type(4)))  int      i32x4;

union HF16 { v16h v; v8h h[2]; };

__device__ __forceinline__ v8f wmma16(v16h a, v16h b, v8f c) {
  return __builtin_amdgcn_wmma_f32_16x16x32_f16(false, a, false, b, (short)0, c,
                                                false, false);
}

// A-fragment (16x32 f16, M rows per lane): lane L<16 -> M=L, K {0..7,16..23};
// lane L>=16 -> M=L-16, K {8..15,24..31}. Source row-major, ld in halves.
__device__ __forceinline__ v16h load_a_frag(const half_t* base, int ld) {
  const int l = threadIdx.x & 31;
  const half_t* p = base + (size_t)(l & 15) * ld + ((l < 16) ? 0 : 8);
  HF16 r;
  r.h[0] = *(const v8h*)(p);
  r.h[1] = *(const v8h*)(p + 16);
  return r.v;
}

// B-fragment (32x16 f16): lane L -> N=L%16; L<16 K {0..7,16..23}, L>=16 K {8..15,24..31}.
__device__ __forceinline__ v16h load_b_frag(const half_t* base, int ld) {
  const int l = threadIdx.x & 31;
  const half_t* p = base + (l & 15) + (size_t)((l < 16) ? 0 : 8) * ld;
  v16h r;
#pragma unroll
  for (int i = 0; i < 8; ++i) {
    r[i]     = p[(size_t)i * ld];
    r[i + 8] = p[(size_t)(i + 16) * ld];
  }
  return r;
}

// A-fragment built from fp32 source with on-the-fly f32->f16 convert.
__device__ __forceinline__ v16h load_a_frag_f32(const float* base, int ld) {
  const int l = threadIdx.x & 31;
  const float* p = base + (size_t)(l & 15) * ld + ((l < 16) ? 0 : 8);
  v16h r;
#pragma unroll
  for (int i = 0; i < 8; ++i) {
    r[i]     = (half_t)p[i];
    r[i + 8] = (half_t)p[i + 16];
  }
  return r;
}

// ---------------------------------------------------------------------------
// TDM: DMA a [64 rows x 64 cols] f16 tile (row stride NTOK halves) from
// global memory into LDS via TENSOR_LOAD_TO_LDS. 2D descriptor, no padding,
// no multicast (workgroup_mask = 0), tracked with TENSORcnt.
// This toolchain exposes the 6-arg builtin: (g0, g1, g2, g3, g4, cpol).
// ---------------------------------------------------------------------------
__device__ __forceinline__ void tdm_load_k_tile(const half_t* gsrc,
                                                unsigned lds_off) {
  const unsigned long long ga = (unsigned long long)(uintptr_t)gsrc;
  u32x4 g0;
  g0[0] = 1u;                                   // count=1, user descriptor
  g0[1] = lds_off;                              // lds_addr (bytes)
  g0[2] = (unsigned)(ga & 0xffffffffu);         // global_addr[31:0]
  g0[3] = (unsigned)((ga >> 32) & 0x1ffffffu)   // global_addr[56:32]
        | (2u << 30);                           // type = 2 ("image")
  i32x8 g1;
  g1[0] = 0x00010000;                           // data_size = 1 -> 2 bytes
  g1[1] = (int)((NTOK & 0xffff) << 16);         // tensor_dim0[15:0]
  g1[2] = (int)((NTOK >> 16) & 0xffff)          // tensor_dim0[31:16]
        | (int)(64u << 16);                     // tensor_dim1[15:0] = 64
  g1[3] = (int)(64u << 16);                     // dim1 hi=0 | tile_dim0 = 64
  g1[4] = 64;                                   // tile_dim1 = 64, tile_dim2 = 0
  g1[5] = NTOK;                                 // tensor_dim0_stride[31:0]
  g1[6] = 0;                                    // stride hi | dim1_stride lo
  g1[7] = 0;
  const i32x4 z4 = {0, 0, 0, 0};                // groups 2/3 unused (2D tile)
  const i32x8 z8 = {0, 0, 0, 0, 0, 0, 0, 0};    // trailing group unused
  __builtin_amdgcn_tensor_load_to_lds(g0, g1, z4, z4, z8, 0);
}

// ---------------------------------------------------------------------------
// Kernel 1: fused projection GEMM  [Wq;Wk;Wd;Wv](704x512) @ x(512xN)
// Outputs (f16, WMMA-friendly layouts):
//   qT [B][N][64], kh [B][64][N], ph [B][64][N], pT [B][N][64], vT [B][N][512]
//   xd [B][64][N] fp32 (channel-attention residual)
// ---------------------------------------------------------------------------
__global__ __launch_bounds__(256)
void k1_proj(const float* __restrict__ x,
             const float* __restrict__ Wq, const float* __restrict__ bq,
             const float* __restrict__ Wk, const float* __restrict__ bk,
             const float* __restrict__ Wv, const float* __restrict__ bv,
             const float* __restrict__ Wd, const float* __restrict__ bd,
             half_t* __restrict__ qT, half_t* __restrict__ kh,
             half_t* __restrict__ ph, half_t* __restrict__ pT,
             half_t* __restrict__ vT, float* __restrict__ xd) {
  __shared__ __attribute__((aligned(64))) half_t xs[32][256];
  __shared__ __attribute__((aligned(64))) half_t wa[16][32];

  const int tid = threadIdx.x, l = tid & 31, w = tid >> 5;
  const int b  = blockIdx.z;
  const int n0 = blockIdx.x * 256;
  const int ot = blockIdx.y;  // 0..43: 4 q-tiles, 4 k-tiles, 4 d-tiles, 32 v-tiles

  const float* Wp; const float* bp; int seg, r0;
  if (ot < 4)       { seg = 0; Wp = Wq; bp = bq; r0 = ot * 16; }
  else if (ot < 8)  { seg = 1; Wp = Wk; bp = bk; r0 = (ot - 4) * 16; }
  else if (ot < 12) { seg = 2; Wp = Wd; bp = bd; r0 = (ot - 8) * 16; }
  else              { seg = 3; Wp = Wv; bp = bv; r0 = (ot - 12) * 16; }

  const float* xb = x + (size_t)b * C_IN * NTOK + n0;

  v8f acc0 = {}, acc1 = {};
  for (int kk = 0; kk < C_IN; kk += 32) {
    __syncthreads();
    {  // stage x tile [32 x 256] fp32 -> f16 into LDS
      const int tr = tid >> 3, tc = (tid & 7) * 32;
      const float* xp = xb + (size_t)(kk + tr) * NTOK + tc;
#pragma unroll
      for (int i = 0; i < 32; ++i) xs[tr][tc + i] = (half_t)xp[i];
    }
    if (tid < 128) {  // stage weight tile [16 x 32]
      const int r = tid >> 3, c4 = (tid & 7) * 4;
      const float* wp = Wp + (size_t)(r0 + r) * C_IN + kk + c4;
#pragma unroll
      for (int i = 0; i < 4; ++i) wa[r][c4 + i] = (half_t)wp[i];
    }
    __syncthreads();
    v16h a = load_a_frag(&wa[0][0], 32);
    acc0 = wmma16(a, load_b_frag(&xs[0][w * 32], 256), acc0);
    acc1 = wmma16(a, load_b_frag(&xs[0][w * 32 + 16], 256), acc1);
  }

  const int rb = 8 * (l >> 4);
#pragma unroll
  for (int s = 0; s < 2; ++s) {
    v8f av = (s == 0) ? acc0 : acc1;
    const int n = n0 + w * 32 + s * 16 + (l & 15);
    float vr[8];
#pragma unroll
    for (int j = 0; j < 8; ++j) vr[j] = av[j] + bp[r0 + rb + j];

    if (seg == 0) {                 // q stored transposed: qT[n][d]
      v8h pk;
#pragma unroll
      for (int j = 0; j < 8; ++j) pk[j] = (half_t)vr[j];
      *(v8h*)(qT + (size_t)b * NTOK * DK + (size_t)n * DK + r0 + rb) = pk;
    } else if (seg == 1) {          // k natural: kh[d][n]
#pragma unroll
      for (int j = 0; j < 8; ++j)
        kh[(size_t)b * DK * NTOK + (size_t)(r0 + rb + j) * NTOK + n] = (half_t)vr[j];
    } else if (seg == 2) {          // p both ways + xd fp32
      v8h pk;
#pragma unroll
      for (int j = 0; j < 8; ++j) pk[j] = (half_t)vr[j];
      *(v8h*)(pT + (size_t)b * NTOK * DK + (size_t)n * DK + r0 + rb) = pk;
#pragma unroll
      for (int j = 0; j < 8; ++j) {
        const size_t o = (size_t)b * DK * NTOK + (size_t)(r0 + rb + j) * NTOK + n;
        ph[o] = (half_t)vr[j];
        xd[o] = vr[j];
      }
    } else {                        // v stored transposed: vT[n][c]
      v8h pk;
#pragma unroll
      for (int j = 0; j < 8; ++j) pk[j] = (half_t)vr[j];
      *(v8h*)(vT + (size_t)b * NTOK * C_IN + (size_t)n * C_IN + r0 + rb) = pk;
    }
  }
}

// ---------------------------------------------------------------------------
// Kernel 2: spatial attention, flash style. Block = 32 query rows x 512 ch.
// 8 waves; wave w owns channels [64w, 64w+64). Online softmax in LDS.
// K tiles DMA'd into LDS by the Tensor Data Mover, double-buffered.
// ---------------------------------------------------------------------------
__global__ __launch_bounds__(256)
void k2_spatial(const float* __restrict__ x, const half_t* __restrict__ qT,
                const half_t* __restrict__ kh, const half_t* __restrict__ vT,
                const float* __restrict__ gamma_s, float* __restrict__ out) {
  __shared__ __attribute__((aligned(64))) half_t ldsK[2][64 * 64];  // TDM dest
  __shared__ __attribute__((aligned(64))) half_t Pl[32 * 64];
  __shared__ float pmax[32][4], psum[32][4];
  __shared__ float runmax[32], runsum[32], rscale[32];

  const int b  = blockIdx.y;
  const int m0 = blockIdx.x * 32;
  const int tid = threadIdx.x, l = tid & 31, w = tid >> 5;
  const int mt = w >> 2, nt = w & 3;
  const int rb = 8 * (l >> 4), col = l & 15;
  const float LOG2E = 1.44269504088896f;
  const int NITER = NTOK / 64;

  const half_t* qTb = qT + (size_t)b * NTOK * DK + (size_t)m0 * DK;
  const half_t* khb = kh + (size_t)b * DK * NTOK;
  const half_t* vTb = vT + (size_t)b * NTOK * C_IN;

  const unsigned ldsK_off0 = (unsigned)(uintptr_t)(&ldsK[0][0]);
  const unsigned ldsK_off1 = (unsigned)(uintptr_t)(&ldsK[1][0]);

  // TDM prologue: DMA first K tile into buffer 0 (wave 0 drives the TDM)
  if (tid < 32) tdm_load_k_tile(khb, ldsK_off0);

  if (tid < 32) { runmax[tid] = -3.0e38f; runsum[tid] = 0.0f; }

  // Q^T fragments for this wave's m-tile (reused every iteration)
  v16h aq0 = load_a_frag(qTb + (size_t)(mt * 16) * DK, DK);
  v16h aq1 = load_a_frag(qTb + (size_t)(mt * 16) * DK + 32, DK);

  v8f zero = {};
  v8f acc[2][4];
#pragma unroll
  for (int i = 0; i < 2; ++i)
#pragma unroll
    for (int j = 0; j < 4; ++j) acc[i][j] = zero;
  __syncthreads();

  for (int it = 0; it < NITER; ++it) {
    const int n0  = it * 64;
    const int cur = it & 1;

    // issue next K tile, then wait for the current one (in-order TENSORcnt)
    if (tid < 32) {
      if (it + 1 < NITER) {
        tdm_load_k_tile(khb + (n0 + 64), cur ? ldsK_off0 : ldsK_off1);
        __builtin_amdgcn_s_wait_tensorcnt(1);
      } else {
        __builtin_amdgcn_s_wait_tensorcnt(0);
      }
    }
    __syncthreads();

    if (n0 + 64 < NTOK)
      __builtin_prefetch(vTb + (size_t)(n0 + 64) * C_IN + w * 64, 0, 1);

    // --- S tile: 16x16 scores, K = 64 over d (K tile from LDS) ---
    v8f s = zero;
    const half_t* kl = &ldsK[cur][0] + nt * 16;
    s = wmma16(aq0, load_b_frag(kl, 64), s);
    s = wmma16(aq1, load_b_frag(kl + 32 * 64, 64), s);

    // --- per-row max across the 16 lanes holding this row's columns ---
    float mx[8];
#pragma unroll
    for (int j = 0; j < 8; ++j) {
      float v = s[j];
#pragma unroll
      for (int off = 1; off <= 8; off <<= 1) v = fmaxf(v, __shfl_xor(v, off, 32));
      mx[j] = v;
    }
    if (col == 0) {
#pragma unroll
      for (int j = 0; j < 8; ++j) pmax[mt * 16 + rb + j][nt] = mx[j];
    }
    __syncthreads();
    if (tid < 32) {
      const float mrow = fmaxf(fmaxf(pmax[tid][0], pmax[tid][1]),
                               fmaxf(pmax[tid][2], pmax[tid][3]));
      const float nm = fmaxf(runmax[tid], mrow);
      rscale[tid] = exp2f((runmax[tid] - nm) * LOG2E);
      runmax[tid] = nm;
    }
    __syncthreads();

    // --- P = exp(S - rowmax); stash to LDS f16; partial row sums ---
    float pv[8];
#pragma unroll
    for (int j = 0; j < 8; ++j) {
      const int r = mt * 16 + rb + j;
      pv[j] = exp2f((s[j] - runmax[r]) * LOG2E);
      Pl[r * 64 + nt * 16 + col] = (half_t)pv[j];
    }
#pragma unroll
    for (int j = 0; j < 8; ++j) {
      float v = pv[j];
#pragma unroll
      for (int off = 1; off <= 8; off <<= 1) v += __shfl_xor(v, off, 32);
      if (col == 0) psum[mt * 16 + rb + j][nt] = v;
    }
    // rescale running output accumulators
#pragma unroll
    for (int mt2 = 0; mt2 < 2; ++mt2) {
      float fs[8];
#pragma unroll
      for (int j = 0; j < 8; ++j) fs[j] = rscale[mt2 * 16 + rb + j];
#pragma unroll
      for (int ct = 0; ct < 4; ++ct)
#pragma unroll
        for (int j = 0; j < 8; ++j) acc[mt2][ct][j] *= fs[j];
    }
    __syncthreads();
    if (tid < 32)
      runsum[tid] = runsum[tid] * rscale[tid] +
                    psum[tid][0] + psum[tid][1] + psum[tid][2] + psum[tid][3];

    // --- O += P @ V^T over this wave's 64-channel slice ---
    v16h p00 = load_a_frag(Pl, 64);
    v16h p01 = load_a_frag(Pl + 32, 64);
    v16h p10 = load_a_frag(Pl + 16 * 64, 64);
    v16h p11 = load_a_frag(Pl + 16 * 64 + 32, 64);
#pragma unroll
    for (int ct = 0; ct < 4; ++ct) {
      const half_t* vb = vTb + (size_t)n0 * C_IN + w * 64 + ct * 16;
      v16h b0 = load_b_frag(vb, C_IN);
      v16h b1 = load_b_frag(vb + (size_t)32 * C_IN, C_IN);
      acc[0][ct] = wmma16(p00, b0, acc[0][ct]);
      acc[0][ct] = wmma16(p01, b1, acc[0][ct]);
      acc[1][ct] = wmma16(p10, b0, acc[1][ct]);
      acc[1][ct] = wmma16(p11, b1, acc[1][ct]);
    }
    __syncthreads();
  }

  // --- epilogue: out = gamma_s * (O / l) + x ---
  const float gs = gamma_s[0];
#pragma unroll
  for (int mt2 = 0; mt2 < 2; ++mt2) {
    float inv[8];
#pragma unroll
    for (int j = 0; j < 8; ++j) inv[j] = 1.0f / runsum[mt2 * 16 + rb + j];
#pragma unroll
    for (int ct = 0; ct < 4; ++ct) {
      const int c = w * 64 + ct * 16 + col;
      const size_t base = (size_t)b * C_IN * NTOK + (size_t)c * NTOK + m0;
#pragma unroll
      for (int j = 0; j < 8; ++j) {
        const int m = mt2 * 16 + rb + j;
        out[base + m] = gs * acc[mt2][ct][j] * inv[j] + x[base + m];
      }
    }
  }
}

// ---------------------------------------------------------------------------
// Kernel 3: channel energy e = p p^T (64x64, K=4096), max-subtract + softmax.
// One block per batch.
// ---------------------------------------------------------------------------
__global__ __launch_bounds__(256)
void k3_chan_attn(const half_t* __restrict__ ph, const half_t* __restrict__ pT,
                  float* __restrict__ cattn) {
  __shared__ __attribute__((aligned(64))) float eL[64][64];
  const int b = blockIdx.x;
  const int tid = threadIdx.x, l = tid & 31, w = tid >> 5;
  const half_t* pb  = ph + (size_t)b * DK * NTOK;
  const half_t* pTb = pT + (size_t)b * NTOK * DK;

  const int mt  = w >> 1;            // tiles 2w, 2w+1 share the row tile
  const int nt0 = (2 * w) & 3, nt1 = nt0 + 1;
  v8f e0 = {}, e1 = {};
  for (int n0 = 0; n0 < NTOK; n0 += 32) {
    v16h a  = load_a_frag(pb + (size_t)(mt * 16) * NTOK + n0, NTOK);
    v16h b0 = load_b_frag(pTb + (size_t)n0 * DK + nt0 * 16, DK);
    v16h b1 = load_b_frag(pTb + (size_t)n0 * DK + nt1 * 16, DK);
    e0 = wmma16(a, b0, e0);
    e1 = wmma16(a, b1, e1);
  }
  const int rb = 8 * (l >> 4), col = l & 15;
#pragma unroll
  for (int j = 0; j < 8; ++j) {
    eL[mt * 16 + rb + j][nt0 * 16 + col] = e0[j];
    eL[mt * 16 + rb + j][nt1 * 16 + col] = e1[j];
  }
  __syncthreads();
  if (tid < 64) {
    float r[64];
    float M = -3.0e38f;
#pragma unroll
    for (int i = 0; i < 64; ++i) { r[i] = eL[tid][i]; M = fmaxf(M, r[i]); }
    float m2 = -3.0e38f;
#pragma unroll
    for (int i = 0; i < 64; ++i) { r[i] = M - r[i]; m2 = fmaxf(m2, r[i]); }
    float sum = 0.0f;
#pragma unroll
    for (int i = 0; i < 64; ++i) { r[i] = __expf(r[i] - m2); sum += r[i]; }
    const float inv = 1.0f / sum;
#pragma unroll
    for (int i = 0; i < 64; ++i)
      cattn[(size_t)b * DK * DK + (size_t)tid * DK + i] = r[i] * inv;
  }
}

// ---------------------------------------------------------------------------
// Kernel 4: channel_out = Wu @ (gamma_c * (c_attn @ p) + xd) + bu; out += it.
// Block = 128 n-columns; phase 1 builds c_out tile in LDS f16; phase 2 GEMMs
// through Wu and accumulates into d_out (stream-ordered after k2).
// ---------------------------------------------------------------------------
__global__ __launch_bounds__(256)
void k4_chan_out(const half_t* __restrict__ ph, const float* __restrict__ xd,
                 const float* __restrict__ cattn, const float* __restrict__ Wu,
                 const float* __restrict__ bu, const float* __restrict__ gamma_c,
                 float* __restrict__ out) {
  __shared__ __attribute__((aligned(64))) half_t cs[64][128];
  const int b = blockIdx.y, n0 = blockIdx.x * 128;
  const int tid = threadIdx.x, l = tid & 31, w = tid >> 5;
  const int rb = 8 * (l >> 4), col = l & 15;
  const float gc = gamma_c[0];
  const half_t* pb  = ph + (size_t)b * DK * NTOK + n0;
  const float*  xdb = xd + (size_t)b * DK * NTOK + n0;
  const float*  cab = cattn + (size_t)b * DK * DK;

  {  // phase 1: c_out[64 x 128] = gamma_c * (c_attn @ p) + xd  -> LDS f16
    const int mt = w & 3, ntb = (w >> 2) * 4;
    v16h a0 = load_a_frag_f32(cab + (size_t)(mt * 16) * DK, DK);
    v16h a1 = load_a_frag_f32(cab + (size_t)(mt * 16) * DK + 32, DK);
    for (int nt = ntb; nt < ntb + 4; ++nt) {
      v8f acc = {};
      acc = wmma16(a0, load_b_frag(pb + nt * 16, NTOK), acc);
      acc = wmma16(a1, load_b_frag(pb + (size_t)32 * NTOK + nt * 16, NTOK), acc);
      const int nc = nt * 16 + col;
#pragma unroll
      for (int j = 0; j < 8; ++j) {
        const int d = mt * 16 + rb + j;
        cs[d][nc] = (half_t)(gc * acc[j] + xdb[(size_t)d * NTOK + nc]);
      }
    }
  }
  __syncthreads();

  // phase 2: out += Wu(512x64) @ c_out(64x128) + bu
  for (int nt = 0; nt < 8; ++nt) {
    v16h b0 = load_b_frag(&cs[0][nt * 16], 128);
    v16h b1 = load_b_frag(&cs[32][nt * 16], 128);
#pragma unroll
    for (int ci = 0; ci < 4; ++ci) {
      const int ct = w * 4 + ci;  // 0..31 -> 512 output channels
      v16h a0 = load_a_frag_f32(Wu + (size_t)(ct * 16) * DK, DK);
      v16h a1 = load_a_frag_f32(Wu + (size_t)(ct * 16) * DK + 32, DK);
      v8f acc = {};
      acc = wmma16(a0, b0, acc);
      acc = wmma16(a1, b1, acc);
      const int nc = n0 + nt * 16 + col;
#pragma unroll
      for (int j = 0; j < 8; ++j) {
        const int c = ct * 16 + rb + j;
        const size_t idx = (size_t)b * C_IN * NTOK + (size_t)c * NTOK + nc;
        out[idx] += acc[j] + bu[c];
      }
    }
  }
}

// ---------------------------------------------------------------------------
extern "C" void kernel_launch(void* const* d_in, const int* in_sizes, int n_in,
                              void* d_out, int out_size, void* d_ws, size_t ws_size,
                              hipStream_t stream) {
  (void)in_sizes; (void)n_in; (void)out_size; (void)ws_size;
  const float* x  = (const float*)d_in[0];
  const float* Wq = (const float*)d_in[1];
  const float* bq = (const float*)d_in[2];
  const float* Wk = (const float*)d_in[3];
  const float* bk = (const float*)d_in[4];
  const float* Wv = (const float*)d_in[5];
  const float* bv = (const float*)d_in[6];
  const float* gs = (const float*)d_in[7];
  const float* Wd = (const float*)d_in[8];
  const float* bd = (const float*)d_in[9];
  const float* Wu = (const float*)d_in[10];
  const float* bu = (const float*)d_in[11];
  const float* gc = (const float*)d_in[12];

  char* ws = (char*)d_ws;
  half_t* qT = (half_t*)(ws);                         //  2 MB  [B][N][64]
  half_t* kh = (half_t*)(ws + (size_t)(2  << 20));    //  2 MB  [B][64][N]
  half_t* ph = (half_t*)(ws + (size_t)(4  << 20));    //  2 MB  [B][64][N]
  half_t* pT = (half_t*)(ws + (size_t)(6  << 20));    //  2 MB  [B][N][64]
  half_t* vT = (half_t*)(ws + (size_t)(8  << 20));    // 16 MB  [B][N][512]
  float*  xd = (float* )(ws + (size_t)(24 << 20));    //  4 MB  [B][64][N]
  float*  ca = (float* )(ws + (size_t)(28 << 20));    // 64 KB  [B][64][64]
  float* out = (float*)d_out;

  k1_proj<<<dim3(16, 44, BATCH), 256, 0, stream>>>(x, Wq, bq, Wk, bk, Wv, bv,
                                                   Wd, bd, qT, kh, ph, pT, vT, xd);
  k2_spatial<<<dim3(NTOK / 32, BATCH), 256, 0, stream>>>(x, qT, kh, vT, gs, out);
  k3_chan_attn<<<dim3(BATCH), 256, 0, stream>>>(ph, pT, ca);
  k4_chan_out<<<dim3(NTOK / 128, BATCH), 256, 0, stream>>>(ph, xd, ca, Wu, bu, gc, out);
}
